// SceneSegCNNTRM_88579405512806
// MI455X (gfx1250) — matrix operand order repaired and checked by reading, compiled, not run on hardware
//
#include <hip/hip_runtime.h>
#include <hip/hip_bf16.h>

typedef __attribute__((ext_vector_type(16))) __bf16 v16bf;
typedef __attribute__((ext_vector_type(8)))  float  v8f;

// ---------------- constants (match reference) ----------------
#define BB    4
#define NN    1024
#define WW    8
#define DD    2048
#define HSZ   256
#define COSC  128
#define DMODEL 640
#define NHEAD 8
#define DHEAD 80
#define FFD   128
#define NLAYER 6
#define MROWS (BB*NN)      // 4096
#define NPADR (NN+2)       // 1026
#define BIGDIV (1<<30)

// ---------------- bf16 helpers (bit-exact, no __bf16 arithmetic) ----------------
__device__ __forceinline__ __bf16 f2bf(float f) {
  unsigned u = __builtin_bit_cast(unsigned, f);
  unsigned r = u + 0x7FFFu + ((u >> 16) & 1u);
  unsigned short h = (unsigned short)(r >> 16);
  return __builtin_bit_cast(__bf16, h);
}
__device__ __forceinline__ float bf2f(__bf16 b) {
  unsigned short h = __builtin_bit_cast(unsigned short, b);
  unsigned u = ((unsigned)h) << 16;
  return __builtin_bit_cast(float, u);
}
__device__ __forceinline__ float wred32(float v) {
  #pragma unroll
  for (int o = 16; o; o >>= 1) v += __shfl_xor(v, o, 32);
  return v;
}

// ---------------- misc prep kernels ----------------
__global__ void zero_bf16_kernel(__bf16* p, long n) {
  long i = (long)blockIdx.x * blockDim.x + threadIdx.x;
  if (i < n) p[i] = __builtin_bit_cast(__bf16, (unsigned short)0);
}

// in: (K rows, C cols) row-major f32 -> out: [c][k] bf16 (column-major for WMMA B)
__global__ void wt_transpose_kernel(const float* __restrict__ in, __bf16* __restrict__ out,
                                    int K, int C) {
  int t = blockIdx.x * blockDim.x + threadIdx.x;
  if (t >= K * C) return;
  int k = t / C, c = t % C;
  out[(long)c * K + k] = f2bf(in[t]);
}

// conv weights (H, D, 3) -> [h][k*D + d] bf16  (matches im2col K-order of the padded rows)
__global__ void conv_w_prep_kernel(const float* __restrict__ in, __bf16* __restrict__ out,
                                   int H, int D) {
  int t = blockIdx.x * blockDim.x + threadIdx.x;
  if (t >= H * D * 3) return;
  int h = t / (D * 3); int r = t - h * D * 3; int d = r / 3; int k = r - d * 3;
  out[(long)h * D * 3 + (long)k * D + d] = f2bf(in[t]);
}

// ---------------- fused soft-attention pooling + cosine branch ----------------
// One block per position. Gram-matrix trick: sim needs only G = X X^T (8x8) + row sums.
__global__ __launch_bounds__(256) void pool_cos_kernel(
    const float* __restrict__ x, const float* __restrict__ attn_w,
    const float* __restrict__ attn_b, const float* __restrict__ cos_w,
    const float* __restrict__ cos_b, __bf16* __restrict__ pooled_pad,
    float* __restrict__ xout) {
  const int idx = blockIdx.x;              // 0..4095
  const int b = idx >> 10, n = idx & 1023;
  const int tid = threadIdx.x, lane = tid & 31, wv = tid >> 5;
  const float* xp = x + (long)idx * WW * DD;

  // acc[0..35] = Gram upper-tri, [36..43] = row sums, [44..51] = attn scores
  float acc[52];
  #pragma unroll
  for (int i = 0; i < 52; ++i) acc[i] = 0.f;

  for (int d = tid; d < DD; d += 256) {
    float aw = attn_w[d];
    float xv[WW];
    #pragma unroll
    for (int w = 0; w < WW; ++w) xv[w] = xp[w * DD + d];
    #pragma unroll
    for (int w = 0; w < WW; ++w) { acc[36 + w] += xv[w]; acc[44 + w] += xv[w] * aw; }
    int p = 0;
    #pragma unroll
    for (int i = 0; i < WW; ++i)
      #pragma unroll
      for (int j = i; j < WW; ++j) acc[p++] += xv[i] * xv[j];
  }

  __shared__ float red[8][52];
  __shared__ float fin[52];
  #pragma unroll
  for (int i = 0; i < 52; ++i) {
    float v = wred32(acc[i]);
    if (lane == 0) red[wv][i] = v;
  }
  __syncthreads();
  if (tid < 52) {
    float s = 0.f;
    #pragma unroll
    for (int w = 0; w < 8; ++w) s += red[w][tid];
    fin[tid] = s;
  }
  __syncthreads();

  // softmax over window (all threads redundantly)
  float ab = attn_b[0];
  float mx = -1e30f;
  #pragma unroll
  for (int w = 0; w < WW; ++w) mx = fmaxf(mx, fin[44 + w] + ab);
  float aww[WW]; float ssum = 0.f;
  #pragma unroll
  for (int w = 0; w < WW; ++w) { aww[w] = expf(fin[44 + w] + ab - mx); ssum += aww[w]; }
  float rs = 1.f / ssum;
  #pragma unroll
  for (int w = 0; w < WW; ++w) aww[w] *= rs;

  // pooled -> bf16 padded row (row b*1026 + n + 1)
  __bf16* prow = pooled_pad + (long)(b * NPADR + n + 1) * DD;
  for (int d = tid; d < DD; d += 256) {
    float s = 0.f;
    #pragma unroll
    for (int w = 0; w < WW; ++w) s += aww[w] * xp[w * DD + d];
    prow[d] = f2bf(s);
  }

  // cosine similarity channels -> xout[:, 512 + c]
  if (tid < COSC) {
    const float* cw = cos_w + tid * 4;
    float bb = cos_b[tid];
    float s1 = 0.f, s2 = 0.f;
    #pragma unroll
    for (int i = 0; i < 4; ++i) { s1 += cw[i] * fin[36 + i]; s2 += cw[i] * fin[36 + 4 + i]; }
    float q11 = 0.f, q22 = 0.f, q12 = 0.f;
    #pragma unroll
    for (int i = 0; i < 4; ++i)
      #pragma unroll
      for (int j = 0; j < 4; ++j) {
        int a1 = i < j ? i : j, b1 = i < j ? j : i;
        float cc = cw[i] * cw[j];
        q11 += cc * fin[a1 * 8 - (a1 * (a1 - 1)) / 2 + (b1 - a1)];
        int a2 = 4 + a1, b2 = 4 + b1;
        q22 += cc * fin[a2 * 8 - (a2 * (a2 - 1)) / 2 + (b2 - a2)];
        int a3 = i, b3 = 4 + j;
        q12 += cc * fin[a3 * 8 - (a3 * (a3 - 1)) / 2 + (b3 - a3)];
      }
    float Df = (float)DD;
    float p12 = q12 + bb * (s1 + s2) + Df * bb * bb;
    float n1 = fmaxf(sqrtf(fmaxf(q11 + 2.f * bb * s1 + Df * bb * bb, 0.f)), 1e-8f);
    float n2 = fmaxf(sqrtf(fmaxf(q22 + 2.f * bb * s2 + Df * bb * bb, 0.f)), 1e-8f);
    xout[(long)idx * DMODEL + 2 * HSZ + tid] = p12 / (n1 * n2);
  }
}

// ---------------- WMMA bf16 GEMM: C = A @ B (+bias, relu, residual) ----------------
// 2x2 register blocking: each wave owns a 32x32 C block (4 accumulators),
// sharing each A fragment across 2 B tiles and vice-versa (1 KB loads / wmma).
// A: row-major bf16, row index remapped rowA += (rowA/rowDiv)*rowMul (conv padded rows).
// Bt: column-major bf16 [col][K]. Fragment layouts per CDNA5 ISA 7.12.2 (wave32).
union Frag2 { v16bf v; uint4 q[2]; };

#define WMMA_BF16(A_, B_, C_) \
  __builtin_amdgcn_wmma_f32_16x16x32_bf16(false, (A_).v, false, (B_).v, (short)0, (C_), false, false)

template<bool BIAS, bool RELU, bool RESID, bool F32OUT, bool BF16OUT>
__global__ __launch_bounds__(256) void gemm_bf16_wmma(
    const __bf16* __restrict__ A, int lda, int rowDiv, int rowMul,
    const __bf16* __restrict__ Bt, int K,
    const float* __restrict__ bias, const float* __restrict__ residIn,
    float* __restrict__ outF, __bf16* __restrict__ outB,
    int ldo, int coloff, int M, int Ncols) {
  const int lane = threadIdx.x & 31;
  const int wid = blockIdx.x * (blockDim.x >> 5) + (threadIdx.x >> 5);
  const int blkM = M >> 5;                 // 32-row blocks
  const int tm = (wid % blkM) << 5;
  const int tn = (wid / blkM) << 5;
  if (tn >= Ncols) return;

  int rowA0 = tm + (lane & 15);
  int rowA1 = rowA0 + 16;
  rowA0 += (rowA0 / rowDiv) * rowMul;
  rowA1 += (rowA1 / rowDiv) * rowMul;
  const int base0 = (lane >> 4) << 3;      // A K-offset: 0 or 8
  const int klo = (lane >> 4) << 3;        // B K-offset: 0 or 8
  const __bf16* pA0 = A + (long)rowA0 * lda + base0;
  const __bf16* pA1 = A + (long)rowA1 * lda + base0;
  const int col0 = tn + (lane & 15);
  const __bf16* pB0 = Bt + (long)col0 * K + klo;
  const __bf16* pB1 = Bt + (long)(col0 + 16) * K + klo;

  v8f acc[2][2] = {{{}, {}}, {{}, {}}};
  for (int k = 0; k < K; k += 32) {
    Frag2 a0, a1, b0, b1;
    a0.q[0] = *(const uint4*)(pA0 + k);
    a0.q[1] = *(const uint4*)(pA0 + k + 16);
    a1.q[0] = *(const uint4*)(pA1 + k);
    a1.q[1] = *(const uint4*)(pA1 + k + 16);
    b0.q[0] = *(const uint4*)(pB0 + k);
    b0.q[1] = *(const uint4*)(pB0 + k + 16);
    b1.q[0] = *(const uint4*)(pB1 + k);
    b1.q[1] = *(const uint4*)(pB1 + k + 16);
    acc[0][0] = WMMA_BF16(a0, b0, acc[0][0]);
    acc[0][1] = WMMA_BF16(a0, b1, acc[0][1]);
    acc[1][0] = WMMA_BF16(a1, b0, acc[1][0]);
    acc[1][1] = WMMA_BF16(a1, b1, acc[1][1]);
  }

  const int mBase = tm + ((lane >> 4) << 3);
  const int nBase = tn + (lane & 15);
  float bv[2] = {0.f, 0.f};
  if (BIAS) { bv[0] = bias[nBase]; bv[1] = bias[nBase + 16]; }
  #pragma unroll
  for (int ti = 0; ti < 2; ++ti)
    #pragma unroll
    for (int tj = 0; tj < 2; ++tj)
      #pragma unroll
      for (int j = 0; j < 8; ++j) {
        int m = mBase + ti * 16 + j;
        int nn = nBase + tj * 16;
        float v = acc[ti][tj][j];
        if (BIAS) v += bv[tj];
        if (RELU) v = fmaxf(v, 0.f);
        long oi = (long)m * ldo + coloff + nn;
        if (RESID) v += residIn[oi];
        if (F32OUT) outF[oi] = v;
        if (BF16OUT) outB[oi] = f2bf(v);
      }
}

// ---------------- conversion helpers ----------------
__global__ void cvt_conv1pad_kernel(const float* __restrict__ xout, __bf16* __restrict__ c1pad) {
  int t = blockIdx.x * blockDim.x + threadIdx.x;
  if (t >= MROWS * HSZ) return;
  int m = t >> 8, c = t & 255;
  int b = m >> 10, n = m & 1023;
  c1pad[(long)(b * NPADR + n + 1) * HSZ + c] = f2bf(xout[(long)m * DMODEL + c]);
}
__global__ void cvt_act_kernel(const float* __restrict__ xout, __bf16* __restrict__ act, long n) {
  long i = (long)blockIdx.x * blockDim.x + threadIdx.x;
  if (i < n) act[i] = f2bf(xout[i]);
}

// ---------------- layernorm (wave per row) -> f32 + bf16 ----------------
__global__ __launch_bounds__(256) void layernorm_kernel(
    const float* __restrict__ in, const float* __restrict__ g, const float* __restrict__ bt,
    float* __restrict__ xout, __bf16* __restrict__ act, int M) {
  int lane = threadIdx.x & 31;
  int row = blockIdx.x * (blockDim.x >> 5) + (threadIdx.x >> 5);
  if (row >= M) return;
  const float* p = in + (long)row * DMODEL;
  float s = 0.f, s2 = 0.f;
  for (int c = lane; c < DMODEL; c += 32) { float v = p[c]; s += v; s2 += v * v; }
  s = wred32(s); s2 = wred32(s2);
  float mean = s * (1.f / DMODEL);
  float var = s2 * (1.f / DMODEL) - mean * mean;
  float rinv = rsqrtf(var + 1e-5f);
  for (int c = lane; c < DMODEL; c += 32) {
    float v = (p[c] - mean) * rinv * g[c] + bt[c];
    xout[(long)row * DMODEL + c] = v;
    act[(long)row * DMODEL + c] = f2bf(v);
  }
}

// ---------------- banded local attention |i-j|<=1 (wave per (pos, head)) ----------------
__global__ __launch_bounds__(256) void band_attn_kernel(
    const float* __restrict__ q, const float* __restrict__ k, const float* __restrict__ v,
    __bf16* __restrict__ ctx) {
  int lane = threadIdx.x & 31;
  int wid = blockIdx.x * (blockDim.x >> 5) + (threadIdx.x >> 5);  // 0..32767
  int h = wid & (NHEAD - 1);
  int pos = wid >> 3;
  int b = pos >> 10, n = pos & 1023;
  const float scale = 0.11180339887498948f;  // 1/sqrt(80)
  const float* qp = q + (long)pos * DMODEL + h * DHEAD;
  float s[3]; bool ok[3];
  #pragma unroll
  for (int jj = 0; jj < 3; ++jj) {
    int j = n - 1 + jj;
    ok[jj] = (j >= 0 && j < NN);
    float d = 0.f;
    if (ok[jj]) {
      const float* kp = k + (long)(b * NN + j) * DMODEL + h * DHEAD;
      for (int t = lane; t < DHEAD; t += 32) d += qp[t] * kp[t];
    }
    s[jj] = wred32(d) * scale;
  }
  float mx = -1e30f;
  #pragma unroll
  for (int jj = 0; jj < 3; ++jj) if (ok[jj]) mx = fmaxf(mx, s[jj]);
  float p[3]; float ssum = 0.f;
  #pragma unroll
  for (int jj = 0; jj < 3; ++jj) { p[jj] = ok[jj] ? expf(s[jj] - mx) : 0.f; ssum += p[jj]; }
  float rs = 1.f / ssum;
  __bf16* cp = ctx + (long)pos * DMODEL + h * DHEAD;
  for (int t = lane; t < DHEAD; t += 32) {
    float a = 0.f;
    #pragma unroll
    for (int jj = 0; jj < 3; ++jj)
      if (ok[jj]) a += p[jj] * v[(long)(b * NN + n - 1 + jj) * DMODEL + h * DHEAD + t];
    cp[t] = f2bf(a * rs);
  }
}

// ---------------- classifier head (Ncols=2: below a WMMA tile; VALU) ----------------
__global__ void cls_kernel(const __bf16* __restrict__ fcact, const float* __restrict__ cw,
                           const float* __restrict__ cb, float* __restrict__ out) {
  int m = blockIdx.x * blockDim.x + threadIdx.x;
  if (m >= MROWS) return;
  const __bf16* p = fcact + (long)m * HSZ;
  float a0 = cb[0], a1 = cb[1];
  for (int kk = 0; kk < HSZ; ++kk) {
    float h = bf2f(p[kk]);
    a0 += h * cw[kk * 2 + 0];
    a1 += h * cw[kk * 2 + 1];
  }
  out[m * 2 + 0] = a0;
  out[m * 2 + 1] = a1;
}

// ---------------- host orchestration ----------------
static inline dim3 gemm_grid(int M, int Ncols) {
  int tiles = (M >> 5) * (Ncols >> 5);   // 32x32 blocks, one per wave
  return dim3((tiles + 7) / 8);
}

extern "C" void kernel_launch(void* const* d_in, const int* in_sizes, int n_in,
                              void* d_out, int out_size, void* d_ws, size_t ws_size,
                              hipStream_t stream) {
  (void)in_sizes; (void)n_in; (void)out_size; (void)ws_size;
  const float* x       = (const float*)d_in[0];
  const float* cos_w   = (const float*)d_in[2];
  const float* cos_b   = (const float*)d_in[3];
  const float* attn_w  = (const float*)d_in[4];
  const float* attn_b  = (const float*)d_in[5];
  const float* conv1_w = (const float*)d_in[6];
  const float* conv1_b = (const float*)d_in[7];
  const float* conv2_w = (const float*)d_in[8];
  const float* conv2_b = (const float*)d_in[9];
  const float* Wq      = (const float*)d_in[10];
  const float* Wk      = (const float*)d_in[11];
  const float* Wv      = (const float*)d_in[12];
  const float* Wo      = (const float*)d_in[13];
  const float* ln1_g   = (const float*)d_in[14];
  const float* ln1_b   = (const float*)d_in[15];
  const float* ff1_w   = (const float*)d_in[16];
  const float* ff1_b   = (const float*)d_in[17];
  const float* ff2_w   = (const float*)d_in[18];
  const float* ff2_b   = (const float*)d_in[19];
  const float* ln2_g   = (const float*)d_in[20];
  const float* ln2_b   = (const float*)d_in[21];
  const float* fc_w    = (const float*)d_in[22];
  const float* fc_b    = (const float*)d_in[23];
  const float* cls_w   = (const float*)d_in[24];
  const float* cls_b   = (const float*)d_in[25];
  float* logits = (float*)d_out;

  char* ws = (char*)d_ws;
  size_t off = 0;
  auto alloc = [&](size_t bytes) -> char* {
    char* p = ws + off;
    off += (bytes + 255) & ~(size_t)255;
    return p;
  };

  __bf16* pooled_pad = (__bf16*)alloc((size_t)BB * NPADR * DD * 2);
  __bf16* c1pad      = (__bf16*)alloc((size_t)BB * NPADR * HSZ * 2);
  float*  xout       = (float*) alloc((size_t)MROWS * DMODEL * 4);
  float*  resid      = (float*) alloc((size_t)MROWS * DMODEL * 4);
  __bf16* act        = (__bf16*)alloc((size_t)MROWS * DMODEL * 2);
  float*  qb         = (float*) alloc((size_t)MROWS * DMODEL * 4);
  float*  kb         = (float*) alloc((size_t)MROWS * DMODEL * 4);
  float*  vb         = (float*) alloc((size_t)MROWS * DMODEL * 4);
  __bf16* ctxb       = (__bf16*)alloc((size_t)MROWS * DMODEL * 2);
  __bf16* ffact      = (__bf16*)alloc((size_t)MROWS * FFD * 2);
  __bf16* fcact      = (__bf16*)alloc((size_t)MROWS * HSZ * 2);
  __bf16* conv1T     = (__bf16*)alloc((size_t)HSZ * 3 * DD * 2);
  __bf16* conv2T     = (__bf16*)alloc((size_t)HSZ * 3 * HSZ * 2);
  __bf16* WqT        = (__bf16*)alloc((size_t)NLAYER * DMODEL * DMODEL * 2);
  __bf16* WkT        = (__bf16*)alloc((size_t)NLAYER * DMODEL * DMODEL * 2);
  __bf16* WvT        = (__bf16*)alloc((size_t)NLAYER * DMODEL * DMODEL * 2);
  __bf16* WoT        = (__bf16*)alloc((size_t)NLAYER * DMODEL * DMODEL * 2);
  __bf16* ff1T       = (__bf16*)alloc((size_t)NLAYER * FFD * DMODEL * 2);
  __bf16* ff2T       = (__bf16*)alloc((size_t)NLAYER * DMODEL * FFD * 2);
  __bf16* fcT        = (__bf16*)alloc((size_t)HSZ * DMODEL * 2);

  // --- weight prep + pad-row zeroing ---
  {
    long n1 = (long)BB * NPADR * DD;
    zero_bf16_kernel<<<(unsigned)((n1 + 255) / 256), 256, 0, stream>>>(pooled_pad, n1);
    long n2 = (long)BB * NPADR * HSZ;
    zero_bf16_kernel<<<(unsigned)((n2 + 255) / 256), 256, 0, stream>>>(c1pad, n2);
    int t1 = HSZ * DD * 3;
    conv_w_prep_kernel<<<(t1 + 255) / 256, 256, 0, stream>>>(conv1_w, conv1T, HSZ, DD);
    int t2 = HSZ * HSZ * 3;
    conv_w_prep_kernel<<<(t2 + 255) / 256, 256, 0, stream>>>(conv2_w, conv2T, HSZ, HSZ);
    int tdm = DMODEL * DMODEL;
    for (int l = 0; l < NLAYER; ++l) {
      wt_transpose_kernel<<<(tdm + 255) / 256, 256, 0, stream>>>(Wq + (long)l * tdm, WqT + (long)l * tdm, DMODEL, DMODEL);
      wt_transpose_kernel<<<(tdm + 255) / 256, 256, 0, stream>>>(Wk + (long)l * tdm, WkT + (long)l * tdm, DMODEL, DMODEL);
      wt_transpose_kernel<<<(tdm + 255) / 256, 256, 0, stream>>>(Wv + (long)l * tdm, WvT + (long)l * tdm, DMODEL, DMODEL);
      wt_transpose_kernel<<<(tdm + 255) / 256, 256, 0, stream>>>(Wo + (long)l * tdm, WoT + (long)l * tdm, DMODEL, DMODEL);
      int tf = DMODEL * FFD;
      wt_transpose_kernel<<<(tf + 255) / 256, 256, 0, stream>>>(ff1_w + (long)l * tf, ff1T + (long)l * tf, DMODEL, FFD);
      wt_transpose_kernel<<<(tf + 255) / 256, 256, 0, stream>>>(ff2_w + (long)l * tf, ff2T + (long)l * tf, FFD, DMODEL);
    }
    int tfc = DMODEL * HSZ;
    wt_transpose_kernel<<<(tfc + 255) / 256, 256, 0, stream>>>(fc_w, fcT, DMODEL, HSZ);
  }

  // --- soft-attention pooling + cosine branch ---
  pool_cos_kernel<<<MROWS, 256, 0, stream>>>(x, attn_w, attn_b, cos_w, cos_b,
                                             pooled_pad, xout);

  // --- conv1 as GEMM over overlapping padded rows: (4096 x 6144) @ (6144 x 256) ---
  gemm_bf16_wmma<true, true, false, true, false>
      <<<gemm_grid(MROWS, HSZ), 256, 0, stream>>>(
          pooled_pad, DD, NN, 2, conv1T, 3 * DD, conv1_b, nullptr,
          xout, nullptr, DMODEL, 0, MROWS, HSZ);
  cvt_conv1pad_kernel<<<(MROWS * HSZ + 255) / 256, 256, 0, stream>>>(xout, c1pad);

  // --- conv2 as GEMM: (4096 x 768) @ (768 x 256) ---
  gemm_bf16_wmma<true, true, false, true, false>
      <<<gemm_grid(MROWS, HSZ), 256, 0, stream>>>(
          c1pad, HSZ, NN, 2, conv2T, 3 * HSZ, conv2_b, nullptr,
          xout, nullptr, DMODEL, HSZ, MROWS, HSZ);

  // --- full activation to bf16 ---
  {
    long n = (long)MROWS * DMODEL;
    cvt_act_kernel<<<(unsigned)((n + 255) / 256), 256, 0, stream>>>(xout, act, n);
  }

  // --- transformer layers ---
  for (int l = 0; l < NLAYER; ++l) {
    long wo = (long)l * DMODEL * DMODEL;
    gemm_bf16_wmma<false, false, false, true, false>
        <<<gemm_grid(MROWS, DMODEL), 256, 0, stream>>>(
            act, DMODEL, BIGDIV, 0, WqT + wo, DMODEL, nullptr, nullptr,
            qb, nullptr, DMODEL, 0, MROWS, DMODEL);
    gemm_bf16_wmma<false, false, false, true, false>
        <<<gemm_grid(MROWS, DMODEL), 256, 0, stream>>>(
            act, DMODEL, BIGDIV, 0, WkT + wo, DMODEL, nullptr, nullptr,
            kb, nullptr, DMODEL, 0, MROWS, DMODEL);
    gemm_bf16_wmma<false, false, false, true, false>
        <<<gemm_grid(MROWS, DMODEL), 256, 0, stream>>>(
            act, DMODEL, BIGDIV, 0, WvT + wo, DMODEL, nullptr, nullptr,
            vb, nullptr, DMODEL, 0, MROWS, DMODEL);

    band_attn_kernel<<<MROWS, 256, 0, stream>>>(qb, kb, vb, ctxb);

    gemm_bf16_wmma<false, false, true, true, false>
        <<<gemm_grid(MROWS, DMODEL), 256, 0, stream>>>(
            ctxb, DMODEL, BIGDIV, 0, WoT + wo, DMODEL, nullptr, xout,
            resid, nullptr, DMODEL, 0, MROWS, DMODEL);
    layernorm_kernel<<<MROWS / 8, 256, 0, stream>>>(
        resid, ln1_g + (long)l * DMODEL, ln1_b + (long)l * DMODEL, xout, act, MROWS);

    gemm_bf16_wmma<true, true, false, false, true>
        <<<gemm_grid(MROWS, FFD), 256, 0, stream>>>(
            act, DMODEL, BIGDIV, 0, ff1T + (long)l * FFD * DMODEL, DMODEL,
            ff1_b + (long)l * FFD, nullptr, nullptr, ffact, FFD, 0, MROWS, FFD);
    gemm_bf16_wmma<true, false, true, true, false>
        <<<gemm_grid(MROWS, DMODEL), 256, 0, stream>>>(
            ffact, FFD, BIGDIV, 0, ff2T + (long)l * DMODEL * FFD, FFD,
            ff2_b + (long)l * DMODEL, xout, resid, nullptr, DMODEL, 0, MROWS, DMODEL);
    layernorm_kernel<<<MROWS / 8, 256, 0, stream>>>(
        resid, ln2_g + (long)l * DMODEL, ln2_b + (long)l * DMODEL, xout, act, MROWS);
  }

  // --- heads ---
  gemm_bf16_wmma<true, true, false, false, true>
      <<<gemm_grid(MROWS, HSZ), 256, 0, stream>>>(
          act, DMODEL, BIGDIV, 0, fcT, DMODEL, fc_b, nullptr,
          nullptr, fcact, HSZ, 0, MROWS, HSZ);
  cls_kernel<<<MROWS / 256, 256, 0, stream>>>(fcact, cls_w, cls_b, logits);
}
